// SessionGraph_29832842838306
// MI455X (gfx1250) — matrix operand order repaired and checked by reading, not compiled, not run on hardware
//
#include <hip/hip_runtime.h>
#include <hip/hip_bf16.h>

typedef __attribute__((ext_vector_type(16))) _Float16 v16h;
typedef __attribute__((ext_vector_type(8)))  _Float16 v8h;
typedef __attribute__((ext_vector_type(8)))  float    v8f;
typedef __attribute__((ext_vector_type(4)))  float    v4f;

// ---- order-preserving float<->uint map for atomicMax-based segment max ----
__device__ __forceinline__ unsigned f32_ord(float x) {
    unsigned u = __float_as_uint(x);
    return (u & 0x80000000u) ? ~u : (u | 0x80000000u);
}
__device__ __forceinline__ float ord_f32(unsigned e) {
    unsigned u = (e & 0x80000000u) ? (e ^ 0x80000000u) : ~e;
    return __uint_as_float(u);
}

__global__ void zero_u32(unsigned* __restrict__ p, long long n) {
    long long i = (long long)blockIdx.x * blockDim.x + threadIdx.x;
    if (i < n) p[i] = 0u;
}

// ---- Phase 1: per-edge logits + segment max (one wave32 per edge) ----
__global__ void edge_logits(const float* __restrict__ h_v, const float* __restrict__ h_d,
                            const float* __restrict__ W_pi, const float* __restrict__ W_M,
                            const int* __restrict__ src1, const int* __restrict__ dst1,
                            float* __restrict__ logits, unsigned* __restrict__ maxU, int E1) {
    int wid  = (blockIdx.x * blockDim.x + threadIdx.x) >> 5;
    int lane = threadIdx.x & 31;
    if (wid >= E1) return;
    int s = src1[wid], d = dst1[wid];
    v4f hs = *(const v4f*)(h_v + (size_t)s * 128 + lane * 4);
    v4f hd = *(const v4f*)(h_v + (size_t)d * 128 + lane * 4);
    v4f he = *(const v4f*)(h_d + (size_t)wid * 128 + lane * 4);
    v4f wp = *(const v4f*)(W_pi + lane * 4);
    v4f w0 = *(const v4f*)(W_M + lane * 4);
    v4f w1 = *(const v4f*)(W_M + 128 + lane * 4);
    float ep = 0.f, mp = 0.f;
#pragma unroll
    for (int j = 0; j < 4; ++j) {
        float uv = hs[j] * hd[j];
        ep += uv * he[j] * wp[j];
        mp += uv * w0[j] + he[j] * w1[j];
    }
#pragma unroll
    for (int off = 16; off; off >>= 1) {
        ep += __shfl_xor(ep, off, 32);
        mp += __shfl_xor(mp, off, 32);
    }
    if (lane == 0) {
        float mask = 1.f / (1.f + expf(-mp));
        float lg = ep * mask;
        logits[wid] = lg;
        atomicMax(maxU + d, f32_ord(lg));
    }
}

// ---- Phase 1: exp + scatter-accumulate (unnormalized) ----
__global__ void edge_accum(const float* __restrict__ h_v, const int* __restrict__ src1,
                           const int* __restrict__ dst1, const float* __restrict__ logits,
                           const unsigned* __restrict__ maxU, float* __restrict__ z,
                           float* __restrict__ ft, int E1) {
    int wid  = (blockIdx.x * blockDim.x + threadIdx.x) >> 5;
    int lane = threadIdx.x & 31;
    if (wid >= E1) return;
    int s = src1[wid], d = dst1[wid];
    float ex = expf(logits[wid] - ord_f32(maxU[d]));
    if (lane == 0) atomicAdd(z + d, ex);
    v4f hv = *(const v4f*)(h_v + (size_t)s * 128 + lane * 4);
    float* dp = ft + (size_t)d * 128 + lane * 4;
#pragma unroll
    for (int j = 0; j < 4; ++j) atomicAdd(dp + j, hv[j] * ex);
}

// ---- Phase 1: node-level softmax normalization ----
__global__ void normalize_ft(float* __restrict__ ft, const float* __restrict__ z, int N) {
    long long i = (long long)blockIdx.x * blockDim.x + threadIdx.x;
    if (i >= (long long)N * 128) return;
    float zz = z[i >> 7];
    ft[i] = (zz > 0.f) ? ft[i] / zz : 0.f;
}

// ---- W[256][128] row-major f32 -> Wt[128][256] column-major f16 ----
__global__ void transpose_w(const float* __restrict__ W, _Float16* __restrict__ Wt) {
    int idx = blockIdx.x * blockDim.x + threadIdx.x;
    if (idx >= 256 * 128) return;
    int n = idx >> 8;   // 0..127 output column
    int k = idx & 255;  // 0..255
    Wt[idx] = (_Float16)W[k * 128 + n];
}

// ============================================================================
// Async staging of the 64KB f16 weight matrix into LDS, chunk-major layout:
// 16B chunk (row, seg)  ->  LDS byte offset  seg*2048 + row*16
//   (row = output column 0..127, seg = k-chunk 0..31, chunk holds k=seg*8..+7)
// B-fragment reads then become  (ldsBase + h*4096 + n*16 + nt*256) + imm(kk):
// affine => DS immediate offsets, and banks 4n mod 64 are conflict-free
// across the 16 lanes of each half-wave.
// Uses gfx1250 GLOBAL_LOAD_ASYNC_TO_LDS_B128 + s_wait_asynccnt.
// Assumes blockDim.x == 256.
// ============================================================================
__device__ __forceinline__ void stage_weights_async(const _Float16* __restrict__ Wg,
                                                    _Float16* __restrict__ Wlds) {
    unsigned ldsBase = (unsigned)(size_t)(void*)Wlds;
    unsigned long long gbase = (unsigned long long)(size_t)(const void*)Wg;
    int tid = threadIdx.x;
#pragma unroll
    for (int i = 0; i < 16; ++i) {
        int c   = tid + i * 256;                        // source 16B chunk (0..4095)
        int row = c >> 5;
        int seg = c & 31;
        unsigned lo = ldsBase + (unsigned)((seg << 11) | (row << 4)); // LDS byte addr
        unsigned go = (unsigned)(c << 4);                             // global byte off
        asm volatile("global_load_async_to_lds_b128 %0, %1, %2"
                     :: "v"(lo), "v"(go), "s"(gbase) : "memory");
    }
    asm volatile("s_wait_asynccnt 0" ::: "memory");
    __syncthreads();
}

// Read one B fragment (32x16 f16) for k-step kk from chunk-major LDS.
// ntBase = Wlds + h*2048 + n*8 + nt*128 (halves); kk folds into DS immediates.
__device__ __forceinline__ v16h load_b(const _Float16* __restrict__ ntBase, int kk) {
    const v8h* p0 = (const v8h*)(ntBase + kk * 4096);
    const v8h* p1 = (const v8h*)(ntBase + kk * 4096 + 1024);
    v8h b0 = *p0;
    v8h b1 = *p1;
    v16h b;
#pragma unroll
    for (int j = 0; j < 8; ++j) {
        b[j]     = b0[j];
        b[8 + j] = b1[j];
    }
    return b;
}

// Load A fragment (16x32 f16 slice) for k-step kk from two f32 row sources.
// Contiguous k order (load_b matches): lane (M=lane&15, h=lane>>4): halves 0..15 = K kb+16h..kb+16h+15
__device__ __forceinline__ v16h load_a_frag(const float* __restrict__ row0,
                                            const float* __restrict__ row1,
                                            int kk, int h) {
    const float* srow = (kk < 4) ? row0 : row1;
    int kb = ((kk < 4) ? kk * 32 : kk * 32 - 128) + 16 * h;
    v4f x0 = *(const v4f*)(srow + kb);
    v4f x1 = *(const v4f*)(srow + kb + 4);
    v4f x2 = *(const v4f*)(srow + kb + 8);
    v4f x3 = *(const v4f*)(srow + kb + 12);
    v16h a;
#pragma unroll
    for (int j = 0; j < 4; ++j) {
        a[j]      = (_Float16)x0[j];
        a[4 + j]  = (_Float16)x1[j];
        a[8 + j]  = (_Float16)x2[j];
        a[12 + j] = (_Float16)x3[j];
    }
    return a;
}

// ---- Phase 2a: f = [h_t | ft[last_idx[t/4]]] @ W_r   (wave grid-strides over 16-row tiles) ----
__global__ void compute_f(const float* __restrict__ h_t, const float* __restrict__ ft,
                          const int* __restrict__ last_idx, const _Float16* __restrict__ Wr16t,
                          float* __restrict__ f, int T) {
    __shared__ __align__(16) _Float16 Wlds[32768];
    stage_weights_async(Wr16t, Wlds);

    int lane = threadIdx.x & 31;
    int n = lane & 15, h = lane >> 4;
    const _Float16* laneBase = Wlds + h * 2048 + n * 8;
    int waveGlobal = (blockIdx.x * blockDim.x + threadIdx.x) >> 5;
    int waveStride = (gridDim.x * blockDim.x) >> 5;
    int tiles = T >> 4;

    for (int tile = waveGlobal; tile < tiles; tile += waveStride) {
        int row = tile * 16 + n;
        const float* arow0 = h_t + (size_t)row * 128;
        const float* arow1 = ft + (size_t)last_idx[row >> 2] * 128;
        v16h A[8];
#pragma unroll
        for (int kk = 0; kk < 8; ++kk) A[kk] = load_a_frag(arow0, arow1, kk, h);

#pragma unroll 1
        for (int nt = 0; nt < 8; ++nt) {
            const _Float16* ntBase = laneBase + nt * 128;
            v8f acc = {0.f, 0.f, 0.f, 0.f, 0.f, 0.f, 0.f, 0.f};
#pragma unroll
            for (int kk = 0; kk < 8; ++kk) {
                v16h B = load_b(ntBase, kk);
                acc = __builtin_amdgcn_wmma_f32_16x16x32_f16(false, A[kk], false, B,
                                                             (short)0, acc, false, false);
            }
            int col = nt * 16 + n;
#pragma unroll
            for (int r = 0; r < 8; ++r)
                f[(size_t)(tile * 16 + r + 8 * h) * 128 + col] = acc[r];
        }
    }
}

// ---- Phase 2b: e2=tanh([ft[src2]|h_p]@W_q); s=dot(e2, f[dst2]); out[dst2]+=ft[src2]*s ----
__global__ void edge_agg(const float* __restrict__ ft, const float* __restrict__ h_p,
                         const _Float16* __restrict__ Wq16t, const float* __restrict__ f,
                         const int* __restrict__ src2, const int* __restrict__ dst2,
                         float* __restrict__ out, int E2) {
    __shared__ __align__(16) _Float16 Wlds[32768];
    stage_weights_async(Wq16t, Wlds);

    int lane = threadIdx.x & 31;
    int n = lane & 15, h = lane >> 4;
    const _Float16* laneBase = Wlds + h * 2048 + n * 8;
    int waveGlobal = (blockIdx.x * blockDim.x + threadIdx.x) >> 5;
    int waveStride = (gridDim.x * blockDim.x) >> 5;
    int tiles = E2 >> 4;

    for (int tile = waveGlobal; tile < tiles; tile += waveStride) {
        int eA = tile * 16 + n;
        const float* arow0 = ft + (size_t)src2[eA] * 128;
        const float* arow1 = h_p + (size_t)eA * 128;
        v16h A[8];
#pragma unroll
        for (int kk = 0; kk < 8; ++kk) A[kk] = load_a_frag(arow0, arow1, kk, h);

        int drow[8];
#pragma unroll
        for (int r = 0; r < 8; ++r) drow[r] = dst2[tile * 16 + r + 8 * h];

        float s[8] = {0.f, 0.f, 0.f, 0.f, 0.f, 0.f, 0.f, 0.f};
#pragma unroll 1
        for (int nt = 0; nt < 8; ++nt) {
            const _Float16* ntBase = laneBase + nt * 128;
            v8f acc = {0.f, 0.f, 0.f, 0.f, 0.f, 0.f, 0.f, 0.f};
#pragma unroll
            for (int kk = 0; kk < 8; ++kk) {
                v16h B = load_b(ntBase, kk);
                acc = __builtin_amdgcn_wmma_f32_16x16x32_f16(false, A[kk], false, B,
                                                             (short)0, acc, false, false);
            }
            int col = nt * 16 + n;
#pragma unroll
            for (int r = 0; r < 8; ++r)
                s[r] += tanhf(acc[r]) * f[(size_t)drow[r] * 128 + col];
        }
        // reduce dot over the 16 lanes sharing half h (rows r+8h live in that half)
#pragma unroll
        for (int off = 1; off < 16; off <<= 1)
#pragma unroll
            for (int r = 0; r < 8; ++r) s[r] += __shfl_xor(s[r], off, 32);

        // scatter: out[dst2[e]] += ft[src2[e]] * s ; lane n covers cols n*8..n*8+7
#pragma unroll
        for (int r = 0; r < 8; ++r) {
            int e = tile * 16 + r + 8 * h;
            const float* sft = ft + (size_t)src2[e] * 128;
            float* orow = out + (size_t)drow[r] * 128;
            float sv = s[r];
            v4f a0 = *(const v4f*)(sft + n * 8);
            v4f a1 = *(const v4f*)(sft + n * 8 + 4);
#pragma unroll
            for (int j = 0; j < 4; ++j) {
                atomicAdd(orow + n * 8 + j, a0[j] * sv);
                atomicAdd(orow + n * 8 + 4 + j, a1[j] * sv);
            }
        }
    }
}

extern "C" void kernel_launch(void* const* d_in, const int* in_sizes, int n_in,
                              void* d_out, int out_size, void* d_ws, size_t ws_size,
                              hipStream_t stream) {
    const float* h_v  = (const float*)d_in[0];
    const float* h_d  = (const float*)d_in[1];
    const float* h_p  = (const float*)d_in[2];
    const float* h_t  = (const float*)d_in[3];
    const float* W_pi = (const float*)d_in[4];
    const float* W_M  = (const float*)d_in[5];
    const float* W_q  = (const float*)d_in[6];
    const float* W_r  = (const float*)d_in[7];
    const int* src1 = (const int*)d_in[8];
    const int* dst1 = (const int*)d_in[9];
    const int* src2 = (const int*)d_in[10];
    const int* dst2 = (const int*)d_in[11];
    const int* last_idx = (const int*)d_in[12];
    float* out = (float*)d_out;

    const int N  = in_sizes[0] / 128;
    const int E1 = in_sizes[8];
    const int E2 = in_sizes[10];
    const int T  = in_sizes[3] / 128;

    // ---- workspace carve (256B aligned) ----
    char* w = (char*)d_ws;
    auto carve = [&](size_t bytes) -> char* {
        char* p = w;
        w += (bytes + 255) & ~(size_t)255;
        return p;
    };
    unsigned*  maxU   = (unsigned*)carve((size_t)N * 4);
    float*     z      = (float*)carve((size_t)N * 4);
    float*     ft     = (float*)carve((size_t)N * 128 * 4);
    float*     logits = (float*)carve((size_t)E1 * 4);
    float*     f      = (float*)carve((size_t)T * 128 * 4);
    _Float16*  Wq16t  = (_Float16*)carve((size_t)128 * 256 * 2);
    _Float16*  Wr16t  = (_Float16*)carve((size_t)128 * 256 * 2);
    (void)ws_size; (void)n_in; (void)out_size;

    const int BT = 256;
    auto nb = [](long long n, int b) { return (int)((n + b - 1) / b); };

    // init
    zero_u32<<<nb(N, BT), BT, 0, stream>>>(maxU, N);
    zero_u32<<<nb(N, BT), BT, 0, stream>>>((unsigned*)z, N);
    zero_u32<<<nb((long long)N * 128, BT), BT, 0, stream>>>((unsigned*)ft, (long long)N * 128);
    zero_u32<<<nb((long long)T * 128, BT), BT, 0, stream>>>((unsigned*)out, (long long)T * 128);

    // phase 1
    edge_logits<<<nb((long long)E1 * 32, BT), BT, 0, stream>>>(h_v, h_d, W_pi, W_M, src1, dst1,
                                                               logits, maxU, E1);
    edge_accum<<<nb((long long)E1 * 32, BT), BT, 0, stream>>>(h_v, src1, dst1, logits, maxU,
                                                              z, ft, E1);
    normalize_ft<<<nb((long long)N * 128, BT), BT, 0, stream>>>(ft, z, N);

    // weight prep
    transpose_w<<<nb(256 * 128, BT), BT, 0, stream>>>(W_q, Wq16t);
    transpose_w<<<nb(256 * 128, BT), BT, 0, stream>>>(W_r, Wr16t);

    // phase 2: grid-stride waves, ~8 tiles per wave to amortize the 64KB LDS staging
    int tilesT  = T / 16;
    int wavesT  = (tilesT + 7) / 8;
    int blocksT = (wavesT + 7) / 8;
    compute_f<<<blocksT, BT, 0, stream>>>(h_t, ft, last_idx, Wr16t, f, T);

    int tilesE  = E2 / 16;
    int wavesE  = (tilesE + 7) / 8;
    int blocksE = (wavesE + 7) / 8;
    edge_agg<<<blocksE, BT, 0, stream>>>(ft, h_p, Wq16t, f, src2, dst2, out, E2);
}
